// MultiHeadAttention_9732395892995
// MI455X (gfx1250) — matrix operand order repaired and checked by reading, not compiled, not run on hardware
//
#include <hip/hip_runtime.h>
#include <hip/hip_bf16.h>

// ---- CDNA5 WMMA types -------------------------------------------------------
typedef __attribute__((ext_vector_type(16))) __bf16        v16bf;
typedef __attribute__((ext_vector_type(8)))  float         v8f;
typedef __attribute__((ext_vector_type(4)))  unsigned int  v4u;

union Frag16 { v4u u[2]; v16bf v; };

__device__ inline v8f v8f_zero() {
    v8f z = {0.f, 0.f, 0.f, 0.f, 0.f, 0.f, 0.f, 0.f};
    return z;
}

// CDNA5 async global->LDS copy (VGLOBAL encoding, tracked by ASYNCcnt).
__device__ inline void async_copy_b128(unsigned int lds_byte, const void* g) {
    asm volatile("global_load_async_to_lds_b128 %0, %1, off"
                 :: "v"(lds_byte), "v"(g) : "memory");
}
__device__ inline void wait_async0() {
    asm volatile("s_wait_asynccnt 0x0" ::: "memory");
}
__device__ inline unsigned int lds_addr32(const void* p) {
    return (unsigned int)(unsigned long long)p;   // LDS offset = low 32 bits
}

// ---- DPP 16-lane row reductions (VALU only, no LDS traffic) -----------------
template <int CTRL>
__device__ inline float dpp_f(float v) {
    return __int_as_float(__builtin_amdgcn_update_dpp(
        0, __float_as_int(v), CTRL, 0xf, 0xf, true));
}
__device__ inline float row_max16(float v) {
    v = fmaxf(v, dpp_f<0xB1>(v));   // quad_perm(1,0,3,2)  : xor 1
    v = fmaxf(v, dpp_f<0x4E>(v));   // quad_perm(2,3,0,1)  : xor 2
    v = fmaxf(v, dpp_f<0x141>(v));  // row_half_mirror     : fold 4s into 8
    v = fmaxf(v, dpp_f<0x140>(v));  // row_mirror          : fold 8s into 16
    return v;
}
__device__ inline float row_sum16(float v) {
    v += dpp_f<0xB1>(v);
    v += dpp_f<0x4E>(v);
    v += dpp_f<0x141>(v);
    v += dpp_f<0x140>(v);
    return v;
}

constexpr int D_MODEL = 1024;
constexpr int SEQ     = 2048;
constexpr int BATCH   = 2;
constexpr int NH      = 16;
constexpr int HD      = 64;
constexpr int MROWS   = BATCH * SEQ; // 4096

// ---- elementwise fp32 -> bf16 ----------------------------------------------
__global__ void cvt_bf16(const float* __restrict__ s, __bf16* __restrict__ d, int n) {
    int i = blockIdx.x * blockDim.x + threadIdx.x;
    if (i < n) d[i] = (__bf16)s[i];
}

// ---- W [in][out] fp32 -> WT [out][in] bf16 ---------------------------------
__global__ void transpose_bf16(const float* __restrict__ W, __bf16* __restrict__ WT) {
    int idx = blockIdx.x * blockDim.x + threadIdx.x; // o*D + i
    int o = idx >> 10;
    int i = idx & (D_MODEL - 1);
    WT[idx] = (__bf16)W[i * D_MODEL + o];
}

// ---- Tiled bf16 WMMA GEMM: C[m][n] = sum_k A[m][k] * BT[n][k] + bias[n] -----
// MODE 0: bf16 out in [B,H,S,Hd] layout (Q/K)
// MODE 1: bf16 out in [B,H,Hd,S] layout (V transposed)
// MODE 2: fp32 out in [M][D] layout (final projection)
template <int MODE>
__global__ __launch_bounds__(256) void proj_gemm(const __bf16* __restrict__ A,
                                                 const __bf16* __restrict__ BT,
                                                 const float*  __restrict__ bias,
                                                 void* __restrict__ outp) {
    __shared__ __align__(16) __bf16 sA[2][128 * 32];
    __shared__ __align__(16) __bf16 sB[2][128 * 32];

    const int tid  = threadIdx.x;
    const int lane = tid & 31;
    const int wave = tid >> 5;           // 8 waves
    const int wm   = wave >> 1;          // 0..3 -> 32-row strip
    const int wn   = wave & 1;           // 0..1 -> 64-col strip
    const int lh   = lane >> 4;          // lane-half
    const int ln   = lane & 15;
    const int m0   = blockIdx.y * 128;
    const int n0   = blockIdx.x * 128;
    const int row  = tid >> 1;           // 0..127
    const int seg  = tid & 1;            // 16-elem (32B) half of a 32-elem row

    // Per-thread staging addresses: 32 bytes of one tile row (2 x b128).
    const __bf16* gA = A  + (size_t)(m0 + row) * D_MODEL + seg * 16;
    const __bf16* gB = BT + (size_t)(n0 + row) * D_MODEL + seg * 16;
    const int ldsOff = (row * 32 + seg * 16);            // element offset in tile

    auto issue_tile = [&](int kt, int buf) {
        const __bf16* a = gA + kt * 32;
        const __bf16* b = gB + kt * 32;
        unsigned int la = lds_addr32(&sA[buf][ldsOff]);
        unsigned int lb = lds_addr32(&sB[buf][ldsOff]);
        async_copy_b128(la,      a);
        async_copy_b128(la + 16, a + 8);
        async_copy_b128(lb,      b);
        async_copy_b128(lb + 16, b + 8);
    };

    v8f acc[2][4];
    #pragma unroll
    for (int i = 0; i < 2; i++)
        #pragma unroll
        for (int j = 0; j < 4; j++) acc[i][j] = v8f_zero();

    constexpr int NT = D_MODEL / 32;
    issue_tile(0, 0);

    for (int kt = 0; kt < NT; ++kt) {
        const int cur = kt & 1;
        wait_async0();        // own async loads into buf[cur] complete
        __syncthreads();      // all waves: buf[cur] full, buf[cur^1] drained
        if (kt + 1 < NT) issue_tile(kt + 1, cur ^ 1);

        // A fragments: 16x32 bf16; lanes 0-15 K 0..7/16..23, lanes 16-31 K 8..15/24..31
        v16bf a[2], b[4];
        #pragma unroll
        for (int i = 0; i < 2; i++) {
            int r = wm * 32 + i * 16 + ln;
            int c = lh * 8;
            Frag16 f;
            f.u[0] = *(const v4u*)(&sA[cur][r * 32 + c]);
            f.u[1] = *(const v4u*)(&sA[cur][r * 32 + c + 16]);
            a[i] = f.v;
        }
        // B fragments: 32x16 bf16; lanes 0-15 K 0..7/16..23, lanes 16-31 K 8..15/24..31
        #pragma unroll
        for (int j = 0; j < 4; j++) {
            int o = wn * 64 + j * 16 + ln;
            int c = lh * 8;
            Frag16 f;
            f.u[0] = *(const v4u*)(&sB[cur][o * 32 + c]);
            f.u[1] = *(const v4u*)(&sB[cur][o * 32 + c + 16]);
            b[j] = f.v;
        }
        #pragma unroll
        for (int i = 0; i < 2; i++)
            #pragma unroll
            for (int j = 0; j < 4; j++)
                acc[i][j] = __builtin_amdgcn_wmma_f32_16x16x32_bf16(
                    false, a[i], false, b[j], (short)0, acc[i][j], false, false);
    }

    // Hoisted bias: one load per n-tile.
    float bcol[4];
    #pragma unroll
    for (int j = 0; j < 4; j++) bcol[j] = bias[n0 + wn * 64 + j * 16 + ln];

    // Epilogue: C layout per 16x16 tile is M = r + 8*lane_half, N = lane&15.
    #pragma unroll
    for (int i = 0; i < 2; i++)
        #pragma unroll
        for (int j = 0; j < 4; j++)
            #pragma unroll
            for (int r = 0; r < 8; r++) {
                int m = m0 + wm * 32 + i * 16 + r + lh * 8;
                int n = n0 + wn * 64 + j * 16 + ln;
                float v = acc[i][j][r] + bcol[j];
                if (MODE == 2) {
                    ((float*)outp)[(size_t)m * D_MODEL + n] = v;
                } else {
                    int bb = m >> 11, s = m & (SEQ - 1);
                    int h  = n >> 6,  d = n & (HD - 1);
                    size_t idx;
                    if (MODE == 0) idx = ((((size_t)bb * NH + h) * SEQ) + s) * HD + d;
                    else           idx = (((size_t)bb * NH + h) * HD + d) * SEQ + s;
                    ((__bf16*)outp)[idx] = (__bf16)v;
                }
            }
}

// ---- Flash-style attention: per wave one 16-row q-tile, 64-key tiles --------
__global__ __launch_bounds__(128) void attn_fwd(const __bf16* __restrict__ Q,
                                                const __bf16* __restrict__ K,
                                                const __bf16* __restrict__ VT,
                                                __bf16* __restrict__ AO) {
    __shared__ __align__(16) __bf16 sP[4][16 * 64];   // per-wave P scratch (2KB)

    const int tid   = threadIdx.x;
    const int lane  = tid & 31;
    const int wave  = tid >> 5;
    const int lh    = lane >> 4;
    const int ln    = lane & 15;
    const int bh    = blockIdx.y;               // b*NH + h
    const int qbase = blockIdx.x * 64 + wave * 16;

    const __bf16* qp = Q  + (size_t)bh * SEQ * HD;
    const __bf16* kp = K  + (size_t)bh * SEQ * HD;
    const __bf16* vp = VT + (size_t)bh * HD * SEQ;

    // Preload Q A-fragments for Hd = 0..31 and 32..63.
    v16bf qa[2];
    {
        const __bf16* rp = qp + (size_t)(qbase + ln) * HD;
        #pragma unroll
        for (int t = 0; t < 2; t++) {
            Frag16 f;
            f.u[0] = *(const v4u*)(rp + t * 32 + lh * 16);
            f.u[1] = *(const v4u*)(rp + t * 32 + lh * 16 + 8);
            qa[t] = f.v;
        }
    }

    float runmax[8], runsum[8];
    #pragma unroll
    for (int r = 0; r < 8; r++) { runmax[r] = -3.0e30f; runsum[r] = 0.f; }
    v8f oacc[4];
    #pragma unroll
    for (int t = 0; t < 4; t++) oacc[t] = v8f_zero();

    for (int kb = 0; kb < SEQ / 64; ++kb) {
        const int kbase = kb * 64;
        v8f sc[4];
        #pragma unroll
        for (int t = 0; t < 4; t++) sc[t] = v8f_zero();

        // scores = Q(16x64) . K^T(64x64): 4 n-tiles x 2 k-steps of WMMA
        #pragma unroll
        for (int ks = 0; ks < 2; ++ks)
            #pragma unroll
            for (int nt = 0; nt < 4; ++nt) {
                const __bf16* krow = kp + (size_t)(kbase + nt * 16 + ln) * HD + ks * 32 + lh * 16;
                Frag16 f;
                f.u[0] = *(const v4u*)(krow);
                f.u[1] = *(const v4u*)(krow + 8);
                sc[nt] = __builtin_amdgcn_wmma_f32_16x16x32_bf16(
                    false, qa[ks], false, f.v, (short)0, sc[nt], false, false);
            }

        // Online softmax: rows live in 16-lane halves of the C layout;
        // cross-lane reductions use DPP (VALU) instead of the LDS pipe.
        #pragma unroll
        for (int r = 0; r < 8; r++) {
            float s[4];
            #pragma unroll
            for (int t = 0; t < 4; t++) s[t] = sc[t][r] * 0.125f;  // 1/sqrt(64)
            float mx = fmaxf(fmaxf(s[0], s[1]), fmaxf(s[2], s[3]));
            mx = row_max16(mx);
            float nm    = fmaxf(runmax[r], mx);
            float alpha = __expf(runmax[r] - nm);
            runmax[r]   = nm;
            float rs = 0.f;
            #pragma unroll
            for (int t = 0; t < 4; t++) {
                float p = __expf(s[t] - nm);
                sc[t][r] = p;
                rs += p;
            }
            rs = row_sum16(rs);
            runsum[r] = runsum[r] * alpha + rs;
            #pragma unroll
            for (int t = 0; t < 4; t++) oacc[t][r] *= alpha;
        }

        // P (C layout, f32) -> per-wave LDS (bf16 [16][64]) -> A fragments.
        __bf16* pw = &sP[wave][0];
        #pragma unroll
        for (int r = 0; r < 8; r++) {
            int m = r + lh * 8;
            #pragma unroll
            for (int t = 0; t < 4; t++)
                pw[m * 64 + t * 16 + ln] = (__bf16)sc[t][r];
        }
        // Same-wave DS producer/consumer: drain DScnt (no cross-wave barrier).
        asm volatile("s_wait_dscnt 0" ::: "memory");
        v16bf pa[2];
        #pragma unroll
        for (int ks = 0; ks < 2; ++ks) {
            Frag16 f;
            f.u[0] = *(const v4u*)(pw + ln * 64 + ks * 32 + lh * 16);
            f.u[1] = *(const v4u*)(pw + ln * 64 + ks * 32 + lh * 16 + 8);
            pa[ks] = f.v;
        }

        // out += P(16x64) . V(64x64); V stored transposed so keys are contiguous.
        #pragma unroll
        for (int t = 0; t < 4; t++) {
            const __bf16* vrow = vp + (size_t)(t * 16 + ln) * SEQ + kbase + lh * 16;
            #pragma unroll
            for (int ks = 0; ks < 2; ++ks) {
                Frag16 f;
                f.u[0] = *(const v4u*)(vrow + ks * 32);
                f.u[1] = *(const v4u*)(vrow + ks * 32 + 8);
                oacc[t] = __builtin_amdgcn_wmma_f32_16x16x32_bf16(
                    false, pa[ks], false, f.v, (short)0, oacc[t], false, false);
            }
        }
    }

    // Normalize and write attention output as bf16 [M][D] for the out-proj GEMM.
    const int b = bh >> 4, h = bh & 15;
    #pragma unroll
    for (int r = 0; r < 8; r++) {
        float inv = 1.0f / runsum[r];
        int q = qbase + r + lh * 8;
        size_t rowoff = ((size_t)(b * SEQ + q)) * D_MODEL + h * HD;
        #pragma unroll
        for (int t = 0; t < 4; t++)
            AO[rowoff + t * 16 + ln] = (__bf16)(oacc[t][r] * inv);
    }
}

// ---- host-side orchestration ------------------------------------------------
extern "C" void kernel_launch(void* const* d_in, const int* in_sizes, int n_in,
                              void* d_out, int out_size, void* d_ws, size_t ws_size,
                              hipStream_t stream) {
    (void)in_sizes; (void)n_in; (void)out_size; (void)ws_size;
    const float* x  = (const float*)d_in[0];
    const float* Wq = (const float*)d_in[1];
    const float* bq = (const float*)d_in[2];
    const float* Wk = (const float*)d_in[3];
    const float* bk = (const float*)d_in[4];
    const float* Wv = (const float*)d_in[5];
    const float* bv = (const float*)d_in[6];
    const float* Wo = (const float*)d_in[7];
    const float* bo = (const float*)d_in[8];

    char* ws = (char*)d_ws;
    size_t off = 0;
    auto carve = [&](size_t bytes) { void* p = ws + off; off += bytes; return p; };
    __bf16* xb  = (__bf16*)carve((size_t)MROWS * D_MODEL * 2);   // 8 MB
    __bf16* wqt = (__bf16*)carve((size_t)D_MODEL * D_MODEL * 2); // 2 MB
    __bf16* wkt = (__bf16*)carve((size_t)D_MODEL * D_MODEL * 2);
    __bf16* wvt = (__bf16*)carve((size_t)D_MODEL * D_MODEL * 2);
    __bf16* wot = (__bf16*)carve((size_t)D_MODEL * D_MODEL * 2);
    __bf16* Qb  = (__bf16*)carve((size_t)MROWS * D_MODEL * 2);   // [B,H,S,Hd]
    __bf16* Kb  = (__bf16*)carve((size_t)MROWS * D_MODEL * 2);   // [B,H,S,Hd]
    __bf16* VTb = (__bf16*)carve((size_t)MROWS * D_MODEL * 2);   // [B,H,Hd,S]
    __bf16* AOb = (__bf16*)carve((size_t)MROWS * D_MODEL * 2);   // [M,D]
    // total workspace: 48 MB

    cvt_bf16<<<(MROWS * D_MODEL) / 256, 256, 0, stream>>>(x, xb, MROWS * D_MODEL);
    transpose_bf16<<<(D_MODEL * D_MODEL) / 256, 256, 0, stream>>>(Wq, wqt);
    transpose_bf16<<<(D_MODEL * D_MODEL) / 256, 256, 0, stream>>>(Wk, wkt);
    transpose_bf16<<<(D_MODEL * D_MODEL) / 256, 256, 0, stream>>>(Wv, wvt);
    transpose_bf16<<<(D_MODEL * D_MODEL) / 256, 256, 0, stream>>>(Wo, wot);

    dim3 gg(D_MODEL / 128, MROWS / 128); // 8 x 32
    proj_gemm<0><<<gg, 256, 0, stream>>>(xb, wqt, bq, Qb);
    proj_gemm<0><<<gg, 256, 0, stream>>>(xb, wkt, bk, Kb);
    proj_gemm<1><<<gg, 256, 0, stream>>>(xb, wvt, bv, VTb);

    attn_fwd<<<dim3(SEQ / 64, BATCH * NH), 128, 0, stream>>>(Qb, Kb, VTb, AOb);

    proj_gemm<2><<<gg, 256, 0, stream>>>(AOb, wot, bo, d_out);
}
